// HierarchicalPretrainedAttention_7971459301544
// MI455X (gfx1250) — matrix operand order repaired and checked by reading, not compiled, not run on hardware
//
#include <hip/hip_runtime.h>
#include <hip/hip_bf16.h>

// ---------------------------------------------------------------------------
// Problem constants (from reference)
// ---------------------------------------------------------------------------
constexpr int BATCH   = 8;
constexpr int SEQ     = 2048;
constexpr int DIM     = 512;     // latent dim == embed dim
constexpr int NCATS   = 1024;
constexpr int NLABELS = 8192;

constexpr int MROWS    = 64;     // query rows per workgroup (attention kernels)
constexpr int SCHUNK   = 128;    // keys processed per flash iteration
constexpr int NTHREADS = 512;    // 16 wave32 waves

// LDS strides (padded to dodge bank conflicts)
constexpr int QLD = DIM + 8;       // bf16 Q tile stride
constexpr int SLD = SCHUNK + 4;    // f32 raw-score tile stride
constexpr int PLD = SCHUNK + 8;    // bf16 probability tile stride

// ---------------------------------------------------------------------------
// WMMA fragment helpers (CDNA5 gfx1250, wave32, 16x16x32 bf16 -> f32)
// ---------------------------------------------------------------------------
typedef __attribute__((ext_vector_type(16))) __bf16 v16bf;
typedef __attribute__((ext_vector_type(8)))  float  v8f;

#define DEVINL __device__ __forceinline__

DEVINL int lane_id() { return (int)(threadIdx.x & 31); }

// A-matrix 16x32 (MxK), contiguous k order used here for A and B:
//   lanes 0-15: M=lane; VGPR v holds K = 2v; lanes 16-31: K += 16
template <typename T>
DEVINL v16bf frag_a(const T* __restrict__ src, int ld) {
  const int l = lane_id();
  const int m = l & 15, h = l >> 4;
  v16bf a;
#pragma unroll
  for (int v = 0; v < 8; ++v) {
    const int k = 2 * v + 16 * h;
    a[2 * v]     = (__bf16)src[m * ld + k];
    a[2 * v + 1] = (__bf16)src[m * ld + k + 1];
  }
  return a;
}

// B-matrix 32x16 (KxN) where B[k][n] = src[n*ld + k]  (i.e. B = src^T)
// Per lane: 16 consecutive 16-bit elements (32B contiguous -> b128 loads)
template <typename T>
DEVINL v16bf frag_b_t(const T* __restrict__ src, int ld) {
  const int l = lane_id();
  const int n = l & 15, h = l >> 4;
  v16bf b;
#pragma unroll
  for (int v = 0; v < 8; ++v) {
    const int k = 2 * v + 16 * h;
    b[2 * v]     = (__bf16)src[n * ld + k];
    b[2 * v + 1] = (__bf16)src[n * ld + k + 1];
  }
  return b;
}

DEVINL v8f wmma_bf16(v16bf a, v16bf b, v8f c) {
  // 8 args: (neg_a, A, neg_b, B, c_mod, C, reuse_a, reuse_b)
  return __builtin_amdgcn_wmma_f32_16x16x32_bf16(false, a, false, b, (short)0, c,
                                                 false, false);
}

// C/D 16x16 f32: VGPR v -> row (v + 8*half), col = lane&15
DEVINL void store_c(float* __restrict__ dst, int ld, v8f c) {
  const int l = lane_id();
  const int n = l & 15, h = l >> 4;
#pragma unroll
  for (int v = 0; v < 8; ++v) dst[(v + 8 * h) * ld + n] = c[v];
}

// ---------------------------------------------------------------------------
// Prep kernel A: f32 -> bf16 copy (K, map_w), 4 elements/thread, vectorized
// ---------------------------------------------------------------------------
__global__ __launch_bounds__(256) void conv_bf16_kernel(
    const float* __restrict__ in, __bf16* __restrict__ out, long n4) {
  const long i = (long)blockIdx.x * blockDim.x + threadIdx.x;
  if (i >= n4) return;
  const float4 v = ((const float4*)in)[i];
  __bf16* o = out + i * 4;
  o[0] = (__bf16)v.x; o[1] = (__bf16)v.y; o[2] = (__bf16)v.z; o[3] = (__bf16)v.w;
}

// ---------------------------------------------------------------------------
// Prep kernel B: V [B][S][D] f32 -> Vt [B][D][S] bf16 (tiled LDS transpose)
// ---------------------------------------------------------------------------
__global__ __launch_bounds__(256) void transpose_bf16_kernel(
    const float* __restrict__ V, __bf16* __restrict__ Vt) {
  __shared__ float tile[32][33];
  const int c = threadIdx.x & 31, r0 = threadIdx.x >> 5;  // 8 rows per pass
  const int d0 = blockIdx.x * 32, s0 = blockIdx.y * 32, b = blockIdx.z;
  const float* Vb = V + ((size_t)b * SEQ + s0) * DIM + d0;
#pragma unroll
  for (int rr = 0; rr < 32; rr += 8)
    tile[rr + r0][c] = Vb[(size_t)(rr + r0) * DIM + c];
  __syncthreads();
  __bf16* Vtb = Vt + ((size_t)b * DIM + d0) * SEQ + s0;
#pragma unroll
  for (int rr = 0; rr < 32; rr += 8)
    Vtb[(size_t)(rr + r0) * SEQ + c] = (__bf16)tile[c][rr + r0];
}

// ---------------------------------------------------------------------------
// Kernel 1: projection  out[m][n] = sum_e A[m][e]*Wbf[n][e] + bias[n]
// block = 256 (8 waves); wave w -> 16x16 tile at (blockIdx.y*16, bx*128+w*16)
// Wbf is pre-converted bf16 (no converts in the contraction loop).
// ---------------------------------------------------------------------------
__global__ __launch_bounds__(256) void proj_kernel(
    const float* __restrict__ A, const __bf16* __restrict__ Wbf,
    const float* __restrict__ bias, float* __restrict__ out) {
  const int tid = threadIdx.x;
  const int m0  = blockIdx.y * 16;
  const int n0  = blockIdx.x * 128 + (tid >> 5) * 16;

  __shared__ __bf16 ldsA[16 * QLD];
  for (int i = tid; i < 16 * DIM; i += 256) {
    const int r = i >> 9, c = i & (DIM - 1);
    ldsA[r * QLD + c] = (__bf16)A[(size_t)(m0 + r) * DIM + c];
  }
  __syncthreads();

  v8f acc = {};
#pragma unroll 4
  for (int kc = 0; kc < DIM / 32; ++kc) {
    v16bf a = frag_a(ldsA + kc * 32, QLD);
    v16bf b = frag_b_t(Wbf + (size_t)n0 * DIM + kc * 32, DIM);  // B = W^T
    acc = wmma_bf16(a, b, acc);
  }
  const float bv = bias[n0 + (lane_id() & 15)];
#pragma unroll
  for (int v = 0; v < 8; ++v) acc[v] += bv;
  store_c(out + (size_t)m0 * DIM + n0, DIM, acc);
}

// ---------------------------------------------------------------------------
// Flash attention kernel (shared by level 1 and level 2).
// 512 threads = 16 waves; wave w: mt = w>>2 (16-row tile), ds = w&3.
//  - Score phase: wave computes 2 score tiles (rows mt, keys [ds*32, ds*32+32))
//  - Softmax: parallel 8-threads-per-row online max/sum; exp emits bf16 P tile
//  - P.V phase: wave owns output slice rows mt, d-cols [ds*128, ds*128+128)
// LEVEL2 adds: Q tile = Q2p[row] + C1[b][cmap[row]]; raw E2 streamed out.
// ---------------------------------------------------------------------------
template <int MTOTAL, bool LEVEL2>
__global__ __launch_bounds__(NTHREADS) void attn_kernel(
    const float* __restrict__ Qp, const float* __restrict__ C1,
    const int* __restrict__ cmap, const __bf16* __restrict__ Kbf,
    const __bf16* __restrict__ Vtb, float* __restrict__ Cdst,
    float* __restrict__ E2dst) {
  const int tid = threadIdx.x;
  const int w  = tid >> 5;
  const int mt = w >> 2;          // 0..3 : 16-row tile
  const int ds = w & 3;           // 0..3 : key group / 128-col d-slice
  const int m0 = blockIdx.x * MROWS;
  const int b  = blockIdx.y;

  const __bf16* Kb   = Kbf + (size_t)b * SEQ * DIM;   // [S][D] bf16
  const __bf16* Vtb_ = Vtb + (size_t)b * DIM * SEQ;   // [D][S] bf16

  __shared__ __bf16 ldsQ[MROWS * QLD];               // bf16 Q tile
  __shared__ float  ldsS[MROWS * SLD];               // raw f32 scores
  __shared__ __bf16 ldsP[MROWS * PLD];               // bf16 probabilities
  __shared__ float  ldsRed[MROWS * 8];
  __shared__ float  rowM[MROWS], rowL[MROWS], rowScale[MROWS];

  // ---- stage Q tile (with level-2 gather-add) ----
  for (int i = tid; i < MROWS * DIM; i += NTHREADS) {
    const int r = i >> 9, c = i & (DIM - 1);
    const int g = m0 + r;
    float q;
    if constexpr (LEVEL2) {
      const int cat = cmap[g];
      q = Qp[(size_t)g * DIM + c] + C1[((size_t)b * NCATS + cat) * DIM + c];
    } else {
      q = Qp[(size_t)g * DIM + c];
    }
    ldsQ[r * QLD + c] = (__bf16)q;
  }
  if (tid < MROWS) { rowM[tid] = -3.0e38f; rowL[tid] = 0.0f; }
  __syncthreads();

  v8f o[8];
#pragma unroll
  for (int t = 0; t < 8; ++t) o[t] = v8f{};

  const int srow = tid >> 3;      // 0..63 : row for softmax phases
  const int spart = tid & 7;      // 0..7  : 16-column slice within row
  const int h = (tid >> 4) & 1;

  for (int s0 = 0; s0 < SEQ; s0 += SCHUNK) {
    // ================= scores: 2 tiles of 16 keys per wave =================
    {
      v8f acc0 = {}, acc1 = {};
      const __bf16* Kt0 = Kb + (size_t)(s0 + ds * 32) * DIM;
      const __bf16* Kt1 = Kt0 + (size_t)16 * DIM;
      const __bf16* Aq  = ldsQ + mt * 16 * QLD;
#pragma unroll 4
      for (int kc = 0; kc < DIM / 32; ++kc) {
        v16bf a  = frag_a(Aq + kc * 32, QLD);
        acc0 = wmma_bf16(a, frag_b_t(Kt0 + kc * 32, DIM), acc0);
        acc1 = wmma_bf16(a, frag_b_t(Kt1 + kc * 32, DIM), acc1);
      }
      store_c(&ldsS[(mt * 16) * SLD + ds * 32], SLD, acc0);
      store_c(&ldsS[(mt * 16) * SLD + ds * 32 + 16], SLD, acc1);
    }
    __syncthreads();

    // ---- read-only phase: stream raw E2 + per-row partial max ----
    if constexpr (LEVEL2) {
      for (int i = tid; i < MROWS * SCHUNK; i += NTHREADS) {
        const int r = i >> 7, c = i & (SCHUNK - 1);
        E2dst[((size_t)b * MTOTAL + m0 + r) * SEQ + s0 + c] = ldsS[r * SLD + c];
      }
    }
    {
      const float* sp = &ldsS[srow * SLD + spart * 16];
      float pm = -3.0e38f;
#pragma unroll
      for (int j = 0; j < 16; ++j) pm = fmaxf(pm, sp[j]);
      ldsRed[srow * 8 + spart] = pm;
    }
    __syncthreads();

    if (tid < MROWS) {
      float mNew = rowM[tid];
#pragma unroll
      for (int p = 0; p < 8; ++p) mNew = fmaxf(mNew, ldsRed[tid * 8 + p]);
      rowScale[tid] = __expf(rowM[tid] - mNew);
      rowM[tid] = mNew;
    }
    __syncthreads();

    // ---- exp: write bf16 P tile, accumulate partial sums ----
    {
      const float mNew = rowM[srow];
      const float* sp = &ldsS[srow * SLD + spart * 16];
      __bf16* pp = &ldsP[srow * PLD + spart * 16];
      float ps = 0.0f;
#pragma unroll
      for (int j = 0; j < 16; ++j) {
        const float p = __expf(sp[j] - mNew);
        pp[j] = (__bf16)p;
        ps += p;
      }
      ldsRed[srow * 8 + spart] = ps;
    }
    __syncthreads();

    if (tid < MROWS) {
      float l = rowL[tid] * rowScale[tid];
#pragma unroll
      for (int p = 0; p < 8; ++p) l += ldsRed[tid * 8 + p];
      rowL[tid] = l;
    }
    __syncthreads();

    // ================= rescale O, accumulate P @ V =================
#pragma unroll
    for (int v = 0; v < 8; ++v) {
      const float sc = rowScale[mt * 16 + v + 8 * h];
#pragma unroll
      for (int t = 0; t < 8; ++t) o[t][v] *= sc;
    }
    const __bf16* Vs = Vtb_ + (size_t)(ds * 128) * SEQ + s0;  // ld = SEQ
#pragma unroll
    for (int kc = 0; kc < SCHUNK / 32; ++kc) {
      v16bf a = frag_a(&ldsP[mt * 16 * PLD + kc * 32], PLD);  // bf16, no cvt
#pragma unroll
      for (int nt = 0; nt < 8; ++nt) {
        v16bf bb = frag_b_t(Vs + (size_t)nt * 16 * SEQ + kc * 32, SEQ);
        o[nt] = wmma_bf16(a, bb, o[nt]);
      }
    }
    __syncthreads();  // LDS tiles reused next iteration
  }

  // ---- finalize: divide by running sum, store output ----
#pragma unroll
  for (int v = 0; v < 8; ++v) {
    const float inv = 1.0f / rowL[mt * 16 + v + 8 * h];
#pragma unroll
    for (int t = 0; t < 8; ++t) o[t][v] *= inv;
  }
  float* Cout = Cdst + ((size_t)b * MTOTAL + m0 + mt * 16) * DIM + ds * 128;
#pragma unroll
  for (int nt = 0; nt < 8; ++nt) store_c(Cout + nt * 16, DIM, o[nt]);
}

// ---------------------------------------------------------------------------
// Launch
// ---------------------------------------------------------------------------
extern "C" void kernel_launch(void* const* d_in, const int* in_sizes, int n_in,
                              void* d_out, int out_size, void* d_ws,
                              size_t ws_size, hipStream_t stream) {
  (void)in_sizes; (void)n_in; (void)out_size; (void)ws_size;

  const float* K    = (const float*)d_in[0];  // [B,S,D]
  const float* V    = (const float*)d_in[1];  // [B,S,D]
  const int*   cmap = (const int*)d_in[2];    // [NLABELS]
  const float* Q1w  = (const float*)d_in[3];  // [NCATS,E]
  const float* Q2w  = (const float*)d_in[4];  // [NLABELS,E]
  const float* mapw = (const float*)d_in[5];  // [D,E]
  const float* mapb = (const float*)d_in[6];  // [D]

  // workspace layout: f32 region then bf16 region
  float* ws  = (float*)d_ws;
  float* Q1p = ws;                                   // NCATS   * DIM f32
  float* Q2p = Q1p + (size_t)NCATS * DIM;            // NLABELS * DIM f32
  float* C1  = Q2p + (size_t)NLABELS * DIM;          // B*NCATS * DIM f32
  __bf16* Kbf = (__bf16*)(C1 + (size_t)BATCH * NCATS * DIM);  // [B,S,D] bf16
  __bf16* Vtb = Kbf + (size_t)BATCH * SEQ * DIM;              // [B,D,S] bf16
  __bf16* Wbf = Vtb + (size_t)BATCH * DIM * SEQ;              // [D,E]   bf16

  float* outC2 = (float*)d_out;                              // [B,NLABELS,D]
  float* outE2 = outC2 + (size_t)BATCH * NLABELS * DIM;      // [B,NLABELS,S]

  // prep: bf16 K, bf16 transposed V, bf16 map_w
  {
    const long n4 = (long)BATCH * SEQ * DIM / 4;
    conv_bf16_kernel<<<dim3((n4 + 255) / 256), 256, 0, stream>>>(K, Kbf, n4);
    transpose_bf16_kernel<<<dim3(DIM / 32, SEQ / 32, BATCH), 256, 0, stream>>>(V, Vtb);
    const long w4 = (long)DIM * DIM / 4;
    conv_bf16_kernel<<<dim3((w4 + 255) / 256), 256, 0, stream>>>(mapw, Wbf, w4);
  }
  // projections
  proj_kernel<<<dim3(DIM / 128, NCATS / 16),   256, 0, stream>>>(Q1w, Wbf, mapb, Q1p);
  proj_kernel<<<dim3(DIM / 128, NLABELS / 16), 256, 0, stream>>>(Q2w, Wbf, mapb, Q2p);
  // level-1: C1 = softmax(Q1p K^T) V
  attn_kernel<NCATS, false><<<dim3(NCATS / MROWS, BATCH), NTHREADS, 0, stream>>>(
      Q1p, nullptr, nullptr, Kbf, Vtb, C1, nullptr);
  // level-2: E2 raw + C2
  attn_kernel<NLABELS, true><<<dim3(NLABELS / MROWS, BATCH), NTHREADS, 0, stream>>>(
      Q2p, C1, cmap, Kbf, Vtb, outC2, outE2);
}
